// VC_31894427140828
// MI455X (gfx1250) — matrix operand order repaired and checked by reading, not compiled, not run on hardware
//
#include <hip/hip_runtime.h>
#include <hip/hip_bf16.h>

typedef __attribute__((ext_vector_type(16))) _Float16 v16h;
typedef __attribute__((ext_vector_type(8)))  float    v8f;

#define NLAYER 5
#define CH 256
#define COUT 80
#define BATCH 8
#define NHEAD 4
#define HDIM 16
#define ALPHA 5.0f

__device__ __forceinline__ float gelu_f(float x) {
    // exact erf-form GELU (matches jax.nn.gelu(approximate=False))
    return 0.5f * x * (1.0f + erff(x * 0.70710678118654752440f));
}

// ---------------------------------------------------------------------------
// Normalize K per head (16-dim L2 norm) into kn buffer [B,64,S].
// ---------------------------------------------------------------------------
__global__ void normk_kernel(const float* __restrict__ kvs, float* __restrict__ kn,
                             int S, int CZ) {
    int i = blockIdx.x * blockDim.x + threadIdx.x;
    int total = BATCH * NHEAD * S;
    if (i >= total) return;
    int b = i / (NHEAD * S);
    int r = i % (NHEAD * S);
    int h = r / S;
    int s = r % S;
    long ibase = ((long)b * (64 + CZ) + h * HDIM) * S + s;
    long obase = ((long)b * 64 + h * HDIM) * S + s;
    float ss = 0.f;
    float vv[HDIM];
#pragma unroll
    for (int c = 0; c < HDIM; ++c) { vv[c] = kvs[ibase + (long)c * S]; ss += vv[c] * vv[c]; }
    float inv = 1.0f / fmaxf(sqrtf(ss), 1e-12f);
#pragma unroll
    for (int c = 0; c < HDIM; ++c) kn[obase + (long)c * S] = vv[c] * inv;
}

// ---------------------------------------------------------------------------
// Cosine attention with online softmax. One thread per (b,h,tq).
// CVH = per-head value channels (cz/4): 16,8,4,2,1 across layers.
// z layout: [B, CZ, T]
// ---------------------------------------------------------------------------
template<int CVH>
__global__ void attn_kernel(const float* __restrict__ qws, const float* __restrict__ kvs,
                            const float* __restrict__ kn, float* __restrict__ z,
                            int T, int S, int CZ, int CW) {
    int tq = blockIdx.x * blockDim.x + threadIdx.x;
    if (tq >= T) return;
    int bh = blockIdx.y;
    int b = bh >> 2;
    int h = bh & 3;

    // load + normalize q
    float qv[HDIM]; float ss = 0.f;
    const float* qb = qws + ((long)b * (64 + CW) + h * HDIM) * T + tq;
#pragma unroll
    for (int c = 0; c < HDIM; ++c) { qv[c] = qb[(long)c * T]; ss += qv[c] * qv[c]; }
    float inv = 1.0f / fmaxf(sqrtf(ss), 1e-12f);
#pragma unroll
    for (int c = 0; c < HDIM; ++c) qv[c] *= inv;

    const float* knb = kn + ((long)b * 64 + h * HDIM) * S;
    const float* vb  = kvs + ((long)b * (64 + CZ) + 64 + h * CVH) * S;

    float m = -1e30f, l = 0.f;
    float acc[CVH];
#pragma unroll
    for (int cc = 0; cc < CVH; ++cc) acc[cc] = 0.f;

    for (int s = 0; s < S; ++s) {
        float dot = 0.f;
#pragma unroll
        for (int c = 0; c < HDIM; ++c) dot += qv[c] * knb[(long)c * S + s];
        float logit = ALPHA * dot;
        float mn = fmaxf(m, logit);
        float scale = __expf(m - mn);
        float p = __expf(logit - mn);
        l = l * scale + p;
#pragma unroll
        for (int cc = 0; cc < CVH; ++cc)
            acc[cc] = acc[cc] * scale + p * vb[(long)cc * S + s];
        m = mn;
    }
    float invl = 1.0f / l;
#pragma unroll
    for (int cc = 0; cc < CVH; ++cc)
        z[((long)b * CZ + h * CVH + cc) * T + tq] = acc[cc] * invl;
}

// ---------------------------------------------------------------------------
// Implicit-GEMM conv (1 or 3 taps, reflect pad for 3) via WMMA f32<-f16.
// Block = 128 threads (4 waves). Block tile = 32(co) x 128(t); each wave
// owns a 32x32 register tile (4 accumulators, 4 WMMAs per 32-deep K chunk;
// A fragments shared across the two t-subtiles, A LDS tile shared by all
// 4 waves). B staged TRANSPOSED [t][k] with padded rows so per-lane
// fragments are contiguous (vector ds loads, conflict-free banks) and
// global activation loads are wave-coalesced along t.
// Epilogues:
//   0: out = acc + bias (+ add)                    [inp / cond / out convs]
//   1: out = gelu(acc+bias) (+ add)                [c3, add = cond result]
//   2: v = add + gelu(acc+bias); upsample x2 to out2 if given, else out [c1]
// ---------------------------------------------------------------------------
template<int TAPS>
__global__ __launch_bounds__(128)
void conv_wmma_kernel(const float* __restrict__ in, long inBStride,
                      const float* __restrict__ w, const float* __restrict__ bias,
                      const float* __restrict__ add,
                      float* __restrict__ out, float* __restrict__ out2,
                      int CI, int CO, int T, int epi) {
    constexpr int AP = 34;                 // padded row length (halves): bank-stride 17
    __shared__ _Float16 Ald[32 * AP];      // A tile: [m 0..31][kk 0..31]
    __shared__ _Float16 Btl[128 * AP];     // B tile transposed: [tt 0..127][kk 0..31]

    const int tid  = threadIdx.x;
    const int lane = tid & 31;
    const int wv   = tid >> 5;             // wave id 0..3
    const int t0   = blockIdx.x * 128;
    const int co0  = blockIdx.y * 32;
    const int b    = blockIdx.z;
    const int K    = CI * TAPS;
    const int tw   = t0 + wv * 32;         // this wave's t base

    v8f acc[4];                            // [ms*2 + jt]
#pragma unroll
    for (int j = 0; j < 4; ++j) acc[j] = (v8f){0.f,0.f,0.f,0.f,0.f,0.f,0.f,0.f};

    const float* inb = in + (long)b * inBStride;

    for (int k0 = 0; k0 < K; k0 += 32) {
        // ---- stage A (weights): 1024 halves, 8 contiguous per thread
        {
            int m   = tid >> 2;            // 0..31
            int kk0 = (tid & 3) * 8;
            int co  = co0 + m;
#pragma unroll
            for (int i = 0; i < 8; ++i) {
                int k = k0 + kk0 + i;
                float val = (co < CO && k < K) ? w[(long)co * K + k] : 0.f;
                Ald[m * AP + kk0 + i] = (_Float16)val;
            }
        }
        // ---- stage B transposed: thread = one t column, sweep k.
        // For fixed kk the 128 threads read consecutive t -> coalesced.
        {
            int tt = tid;
#pragma unroll
            for (int kk = 0; kk < 32; ++kk) {
                int k = k0 + kk;
                float val = 0.f;
                if (k < K) {
                    int ci  = k / TAPS;
                    int tap = k % TAPS;
                    int tg  = t0 + tt + tap - (TAPS / 2);
                    if (TAPS > 1) {                    // reflect pad (1,1)
                        if (tg < 0)  tg = -tg;
                        if (tg >= T) tg = 2 * T - 2 - tg;
                    }
                    val = inb[(long)ci * T + tg];
                }
                Btl[tt * AP + kk] = (_Float16)val;
            }
        }
        // prefetch next K-chunk's activation rows (global_prefetch path)
        if (k0 + 32 < K) {
            int kpf = k0 + 32 + lane;
            if (kpf < K) {
                int ci = kpf / TAPS;
                __builtin_prefetch(inb + (long)ci * T + t0 + (tid >> 5) * 32, 0, 3);
            }
        }
        __syncthreads();

        // ---- A fragments (two co sub-tiles), contiguous 8-half runs
        const int mA = lane & 15;
        const int kb = (lane & 16) ? 16 : 0;
        v16h af[2];
#pragma unroll
        for (int ms = 0; ms < 2; ++ms) {
            const _Float16* ar = &Ald[(ms * 16 + mA) * AP];
#pragma unroll
            for (int i = 0; i < 8; ++i) {
                af[ms][i]     = ar[kb + i];
                af[ms][i + 8] = ar[kb + 8 + i];
            }
        }
        // ---- B fragments (two t sub-tiles) + 4 WMMAs
        const int nB = lane & 15;
        const int ks = (lane & 16) ? 16 : 0;
#pragma unroll
        for (int jt = 0; jt < 2; ++jt) {
            v16h bf;
            const _Float16* br = &Btl[(wv * 32 + jt * 16 + nB) * AP + ks];
#pragma unroll
            for (int i = 0; i < 16; ++i) bf[i] = br[i];
#pragma unroll
            for (int ms = 0; ms < 2; ++ms) {
                acc[ms * 2 + jt] = __builtin_amdgcn_wmma_f32_16x16x32_f16(
                    false, af[ms], false, bf, (short)0, acc[ms * 2 + jt], false, false);
            }
        }
        __syncthreads();
    }

    // ---- epilogue; C layout: VGPR r -> M = r (+8 if lane>=16), N = lane&15
    {
        const int nC    = lane & 15;
        const int mbase = (lane & 16) ? 8 : 0;
#pragma unroll
        for (int ms = 0; ms < 2; ++ms) {
#pragma unroll
            for (int jt = 0; jt < 2; ++jt) {
                int t = tw + jt * 16 + nC;
#pragma unroll
                for (int r = 0; r < 8; ++r) {
                    int co = co0 + ms * 16 + mbase + r;
                    if (co >= CO) continue;
                    float v = acc[ms * 2 + jt][r] + bias[co];
                    long idx = ((long)b * CO + co) * T + t;
                    if (epi == 0) {
                        if (add) v += add[idx];
                        out[idx] = v;
                    } else if (epi == 1) {
                        v = gelu_f(v);
                        if (add) v += add[idx];
                        out[idx] = v;
                    } else {
                        v = add[idx] + gelu_f(v);
                        if (out2) {
                            long o2 = ((long)b * CO + co) * (2 * (long)T) + 2 * (long)t;
                            out2[o2]     = v;
                            out2[o2 + 1] = v;
                        } else {
                            out[idx] = v;
                        }
                    }
                }
            }
        }
    }
}

// ---------------------------------------------------------------------------
extern "C" void kernel_launch(void* const* d_in, const int* in_sizes, int n_in,
                              void* d_out, int out_size, void* d_ws, size_t ws_size,
                              hipStream_t stream) {
    (void)in_sizes; (void)n_in; (void)out_size; (void)ws_size;

    const int Tn[NLAYER]  = {128, 256, 512, 1024, 2048};
    const int Sn[NLAYER]  = {64, 128, 256, 512, 1024};
    const int CZn[NLAYER] = {64, 32, 16, 8, 4};   // cz == cw per layer

    const float* qws[NLAYER];
    const float* kvs[NLAYER];
    for (int n = 0; n < NLAYER; ++n) {
        qws[n] = (const float*)d_in[n];
        kvs[n] = (const float*)d_in[5 + n];
    }
    auto P = [&](int i) { return (const float*)d_in[10 + i]; };

    // workspace layout (floats)
    float* ws   = (float*)d_ws;
    const long RSZ = (long)BATCH * CH * 2048;     // 4,194,304
    float* RS0  = ws;
    float* RS1  = RS0 + RSZ;
    float* SKIP = RS1 + RSZ;
    float* ZB   = SKIP + RSZ;                     // max 65536
    float* KN   = ZB + 65536;                     // max B*64*1024 = 524288

    float* rs_cur  = RS0;
    float* rs_next = RS1;

    for (int n = 0; n < NLAYER; ++n) {
        const int T = Tn[n], S = Sn[n], CZ = CZn[n], CW = CZn[n];

        // --- attention: normalize k, then online-softmax attention -> ZB
        {
            int nk = BATCH * NHEAD * S;
            normk_kernel<<<(nk + 255) / 256, 256, 0, stream>>>(kvs[n], KN, S, CZ);
            dim3 ag((T + 255) / 256, BATCH * NHEAD);
            switch (CZ / NHEAD) {
            case 16: attn_kernel<16><<<ag, 256, 0, stream>>>(qws[n], kvs[n], KN, ZB, T, S, CZ, CW); break;
            case 8:  attn_kernel<8><<<ag, 256, 0, stream>>>(qws[n], kvs[n], KN, ZB, T, S, CZ, CW); break;
            case 4:  attn_kernel<4><<<ag, 256, 0, stream>>>(qws[n], kvs[n], KN, ZB, T, S, CZ, CW); break;
            case 2:  attn_kernel<2><<<ag, 256, 0, stream>>>(qws[n], kvs[n], KN, ZB, T, S, CZ, CW); break;
            default: attn_kernel<1><<<ag, 256, 0, stream>>>(qws[n], kvs[n], KN, ZB, T, S, CZ, CW); break;
            }
        }

        const float* wInp  = P(n * 8 + 0); const float* bInp  = P(n * 8 + 1);
        const float* wCond = P(n * 8 + 2); const float* bCond = P(n * 8 + 3);
        const float* wC3   = P(n * 8 + 4); const float* bC3   = P(n * 8 + 5);
        const float* wC1   = P(n * 8 + 6); const float* bC1   = P(n * 8 + 7);

        dim3 g(T / 128, CH / 32, BATCH);

        // rs = (rs +) inp_conv1x1(w-part of qws)
        conv_wmma_kernel<1><<<g, 128, 0, stream>>>(
            qws[n] + (long)64 * T, (long)(64 + CW) * T, wInp, bInp,
            (n ? rs_cur : nullptr), rs_cur, nullptr, CW, CH, T, 0);

        // SKIP = cond_conv1x1(z) + bias
        conv_wmma_kernel<1><<<g, 128, 0, stream>>>(
            ZB, (long)CZ * T, wCond, bCond,
            nullptr, SKIP, nullptr, CZ, CH, T, 0);

        // SKIP = gelu(conv3(rs)+b3) + SKIP      (reflect-padded 3-tap)
        conv_wmma_kernel<3><<<g, 128, 0, stream>>>(
            rs_cur, (long)CH * T, wC3, bC3,
            SKIP, SKIP, nullptr, CH, CH, T, 1);

        // rs = rs + gelu(conv1(SKIP)+b1); upsample x2 into rs_next for n<4
        conv_wmma_kernel<1><<<g, 128, 0, stream>>>(
            SKIP, (long)CH * T, wC1, bC1,
            rs_cur, rs_cur, (n < NLAYER - 1 ? rs_next : nullptr), CH, CH, T, 2);

        if (n < NLAYER - 1) { float* t = rs_cur; rs_cur = rs_next; rs_next = t; }
    }

    // out = conv1x1(rs, out_w) + out_b  -> [8, 80, 2048]
    {
        dim3 go(2048 / 128, (COUT + 31) / 32, BATCH);
        conv_wmma_kernel<1><<<go, 128, 0, stream>>>(
            rs_cur, (long)CH * 2048, (const float*)d_in[50], (const float*)d_in[51],
            nullptr, (float*)d_out, nullptr, CH, COUT, 2048, 0);
    }
}
